// TransformerBlock_75256416960603
// MI455X (gfx1250) — compile-verified, matrix operand order checked by reading, not run on hardware
//
#include <hip/hip_runtime.h>
#include <hip/hip_bf16.h>

// ---------------------------------------------------------------------------
// TransformerBlock for MI455X (gfx1250, wave32, WMMA)
//   attn[b]   = softmax(Q K^T / 8) V          (B=2048, T=64, E=256, f32 in/out)
//   out       = relu((attn+query).reshape(B,16384) @ W_ff^T + b_ff)
// Memory-bound (~0.8 GB @ 23.3 TB/s); matmuls in bf16 WMMA w/ f32 accumulate.
// ---------------------------------------------------------------------------

typedef __attribute__((ext_vector_type(16))) __bf16 v16bf;
typedef __attribute__((ext_vector_type(8)))  float  v8f;

#define T_CTX   64
#define EMB     256
#define N_BATCH 2048
#define FF_OUT  256
#define FF_IN   (T_CTX * EMB)   // 16384
#define KCHUNK  128

// Load a 16x32 bf16 fragment in the ISA "A-matrix 16x32" register layout from
// a row-major bf16 matrix in LDS. Used for both A (rows of left operand) and
// B (rows of the transposed right operand == columns of B).
//   lanes 0-15 : M = lane,    v[0..7] = K k0+0..7,  v[8..15] = K k0+16..23
//   lanes 16-31: M = lane-16, v[0..7] = K k0+8..15, v[8..15] = K k0+24..31
__device__ __forceinline__ v16bf load_frag16x32(const __bf16* __restrict__ src,
                                                int ld, int row0, int k0, int lane) {
  const int m  = row0 + (lane & 15);
  const int kb = k0 + ((lane >> 4) << 3);
  const __bf16* p = src + m * ld + kb;
  v16bf f;
#pragma unroll
  for (int e = 0; e < 8; ++e) f[e] = p[e];
#pragma unroll
  for (int e = 0; e < 8; ++e) f[8 + e] = p[16 + e];
  return f;
}

// ---------------------------------------------------------------------------
// Kernel 1: full attention per batch. 1 workgroup (8 wave32) per batch.
// ---------------------------------------------------------------------------
__global__ void __launch_bounds__(256)
attn_kernel(const float* __restrict__ value, const float* __restrict__ key,
            const float* __restrict__ query, float* __restrict__ attn) {
  __shared__ __bf16 sQ [T_CTX * EMB];    // 32 KB
  __shared__ __bf16 sK [T_CTX * EMB];    // 32 KB
  __shared__ __bf16 sVT[EMB * T_CTX];    // 32 KB (V transposed: row n = V[:,n])
  __shared__ float  sS [T_CTX * T_CTX];  // 16 KB scores
  __shared__ __bf16 sW [T_CTX * T_CTX];  //  8 KB softmax weights

  const int b    = blockIdx.x;
  const int tid  = threadIdx.x;
  const int lane = tid & 31;
  const int wave = tid >> 5;

  const float* Qg = query + (size_t)b * T_CTX * EMB;
  const float* Kg = key   + (size_t)b * T_CTX * EMB;
  const float* Vg = value + (size_t)b * T_CTX * EMB;

  // Stage f32 -> bf16 into LDS (V stored transposed for contiguous B-fragments)
  for (int i = tid; i < T_CTX * EMB; i += 256) {
    sQ[i] = (__bf16)Qg[i];
    sK[i] = (__bf16)Kg[i];
    const int r = i >> 8;          // i / EMB
    const int c = i & (EMB - 1);   // i % EMB
    sVT[c * T_CTX + r] = (__bf16)Vg[i];
  }
  __syncthreads();

  // S = (Q K^T) * 1/sqrt(64): 4x4 tiles of 16x16, 2 tiles per wave
  for (int t = wave; t < 16; t += 8) {
    const int ti = t >> 2, tj = t & 3;
    v8f acc = {};
#pragma unroll
    for (int kk = 0; kk < EMB; kk += 32) {
      v16bf a  = load_frag16x32(sQ, EMB, ti * 16, kk, lane);
      v16bf bf = load_frag16x32(sK, EMB, tj * 16, kk, lane);
      acc = __builtin_amdgcn_wmma_f32_16x16x32_bf16(false, a, false, bf,
                                                    (short)0, acc, false, false);
    }
    const int n  = lane & 15;
    const int mb = (lane >> 4) << 3;
#pragma unroll
    for (int r = 0; r < 8; ++r)
      sS[(ti * 16 + mb + r) * T_CTX + tj * 16 + n] = acc[r] * 0.125f;
  }
  __syncthreads();

  // Row softmax (64 rows, one thread each), store weights as bf16
  if (tid < T_CTX) {
    float* row = sS + tid * T_CTX;
    float mx = -3.0e38f;
#pragma unroll 8
    for (int j = 0; j < T_CTX; ++j) mx = fmaxf(mx, row[j]);
    float sum = 0.0f;
#pragma unroll 8
    for (int j = 0; j < T_CTX; ++j) { float e = __expf(row[j] - mx); row[j] = e; sum += e; }
    const float inv = 1.0f / sum;
#pragma unroll 8
    for (int j = 0; j < T_CTX; ++j) sW[tid * T_CTX + j] = (__bf16)(row[j] * inv);
  }
  __syncthreads();

  // A = W V : 4x16 tiles of 16x16, 8 tiles per wave; write f32 to global
  float* Ag = attn + (size_t)b * T_CTX * EMB;
  for (int t = wave; t < 64; t += 8) {
    const int ti = t >> 4, tj = t & 15;
    v8f acc = {};
#pragma unroll
    for (int kk = 0; kk < T_CTX; kk += 32) {
      v16bf a  = load_frag16x32(sW,  T_CTX, ti * 16, kk, lane);
      v16bf bf = load_frag16x32(sVT, T_CTX, tj * 16, kk, lane);
      acc = __builtin_amdgcn_wmma_f32_16x16x32_bf16(false, a, false, bf,
                                                    (short)0, acc, false, false);
    }
    const int n  = lane & 15;
    const int mb = (lane >> 4) << 3;
#pragma unroll
    for (int r = 0; r < 8; ++r)
      Ag[(ti * 16 + mb + r) * EMB + tj * 16 + n] = acc[r];
  }
}

// ---------------------------------------------------------------------------
// Kernel 2: out = relu((attn+query) @ W_ff^T + b_ff)
// [2048 x 16384] x [16384 x 256]; 16-batch M-tile per block, K in 128 chunks.
// ---------------------------------------------------------------------------
__global__ void __launch_bounds__(256)
ff_kernel(const float* __restrict__ attn, const float* __restrict__ query,
          const float* __restrict__ Wff,  const float* __restrict__ bff,
          float* __restrict__ out) {
  __shared__ __bf16 sX [16     * KCHUNK];  //  4 KB: X chunk (attn + query)
  __shared__ __bf16 sWt[FF_OUT * KCHUNK];  // 64 KB: W chunk (row j = B column j)

  const int tid  = threadIdx.x;
  const int lane = tid & 31;
  const int wave = tid >> 5;
  const int m0   = blockIdx.x * 16;  // batch tile base
  const int n0   = wave * 2;         // this wave's two 16-wide N tiles

  v8f acc0 = {};
  v8f acc1 = {};

  for (int k0 = 0; k0 < FF_IN; k0 += KCHUNK) {
    // X = attn + query (fused residual), f32 -> bf16
    for (int i = tid; i < 16 * KCHUNK; i += 256) {
      const int r = i >> 7, c = i & (KCHUNK - 1);
      const size_t gi = (size_t)(m0 + r) * FF_IN + k0 + c;
      sX[i] = (__bf16)(attn[gi] + query[gi]);
    }
    // W_ff chunk: rows are contiguous -> exactly the B-fragment layout we need
    for (int i = tid; i < FF_OUT * KCHUNK; i += 256) {
      const int r = i >> 7, c = i & (KCHUNK - 1);
      sWt[i] = (__bf16)Wff[(size_t)r * FF_IN + k0 + c];
    }
    // Hint the next W chunk toward L2 while we compute (global_prefetch_b8)
    if (k0 + KCHUNK < FF_IN)
      __builtin_prefetch(&Wff[(size_t)tid * FF_IN + k0 + KCHUNK], 0, 1);
    __syncthreads();

#pragma unroll
    for (int kk = 0; kk < KCHUNK; kk += 32) {
      v16bf a  = load_frag16x32(sX, KCHUNK, 0, kk, lane);
      v16bf b0 = load_frag16x32(sWt, KCHUNK, n0 * 16, kk, lane);
      v16bf b1 = load_frag16x32(sWt, KCHUNK, (n0 + 1) * 16, kk, lane);
      acc0 = __builtin_amdgcn_wmma_f32_16x16x32_bf16(false, a, false, b0,
                                                     (short)0, acc0, false, false);
      acc1 = __builtin_amdgcn_wmma_f32_16x16x32_bf16(false, a, false, b1,
                                                     (short)0, acc1, false, false);
    }
    __syncthreads();
  }

  // Bias + ReLU + store
  const int n  = lane & 15;
  const int mb = (lane >> 4) << 3;
  const float bias0 = bff[n0 * 16 + n];
  const float bias1 = bff[(n0 + 1) * 16 + n];
#pragma unroll
  for (int r = 0; r < 8; ++r) {
    const int m = m0 + mb + r;
    out[(size_t)m * FF_OUT + n0 * 16 + n]       = fmaxf(acc0[r] + bias0, 0.0f);
    out[(size_t)m * FF_OUT + (n0 + 1) * 16 + n] = fmaxf(acc1[r] + bias1, 0.0f);
  }
}

// ---------------------------------------------------------------------------
extern "C" void kernel_launch(void* const* d_in, const int* in_sizes, int n_in,
                              void* d_out, int out_size, void* d_ws, size_t ws_size,
                              hipStream_t stream) {
  (void)in_sizes; (void)n_in; (void)out_size; (void)d_ws; (void)ws_size;

  const float* value = (const float*)d_in[0];
  const float* key_  = (const float*)d_in[1];
  const float* query = (const float*)d_in[2];
  // d_in[3] = mask (unused by the block)
  const float* Wff   = (const float*)d_in[4];
  const float* bff   = (const float*)d_in[5];

  float* out  = (float*)d_out;                                   // [2048, 256]
  float* attn = (float*)d_out + (size_t)N_BATCH * FF_OUT;        // [2048, 64, 256]

  attn_kernel<<<N_BATCH, 256, 0, stream>>>(value, key_, query, attn);
  ff_kernel<<<N_BATCH / 16, 256, 0, stream>>>(attn, query, Wff, bff, out);
}
